// FP8MoELayer_31456340476143
// MI455X (gfx1250) — matrix operand order repaired and checked by reading, not compiled, not run on hardware
//
#include <hip/hip_runtime.h>
#include <hip/hip_bf16.h>

// ---------------------------------------------------------------------------
// FP8 MoE layer for gfx1250 (MI455X), f16 WMMA path.
// Double-buffered LDS + register-staged software pipeline:
//   global loads (s+1) -> WMMA (s) -> cvt+LDS store (s+1) -> barrier.
// T=8192 tokens, D=2048 hidden, E=16 experts, top-k=4, F=1408, FS=2048.
// ---------------------------------------------------------------------------

#define T_TOK 8192
#define DDIM  2048
#define NEXP  16
#define FDIM  1408
#define FSDIM 2048

#define TILE_M 128
#define TILE_N 64
#define TILE_K 32
#define KPAD   8   // halves; row stride = 40 halves = 80 B (16 B aligned)

typedef __attribute__((ext_vector_type(16))) _Float16 v16h;
typedef __attribute__((ext_vector_type(8)))  _Float16 v8h;
typedef __attribute__((ext_vector_type(4)))  _Float16 v4h;
typedef __attribute__((ext_vector_type(8)))  float    v8f;

__device__ inline v16h cat8(v8h lo, v8h hi) {
    return __builtin_shufflevector(lo, hi,
        0, 1, 2, 3, 4, 5, 6, 7, 8, 9, 10, 11, 12, 13, 14, 15);
}

// A fragment (16x32 f16, ISA 7.12.2): lane holds row M=lane&15;
// halves [kh..kh+7] and [kh+16..kh+23], kh = 8*(lane>>4).  -> 2x ds_load_b128
__device__ inline v16h load_frag_a(const _Float16 (*As)[TILE_K + KPAD],
                                   int m, int lane)
{
    const int kh = (lane >> 4) * 8;
    v8h lo = *(const v8h*)&As[m][kh];
    v8h hi = *(const v8h*)&As[m][kh + 16];
    return cat8(lo, hi);
}

// B fragment (32x16 f16): lane holds col N=lane&15; K = [kh..kh+7] and [kh+16..kh+23], kh = 8*(lane>>4).
// Staged TRANSPOSED (Bt[n][k]) so the fragment is contiguous -> 2x ds_load_b128
__device__ inline v16h load_frag_bt(const _Float16 (*Bt)[TILE_K + KPAD],
                                    int n, int lane)
{
    const int bk = (lane >> 4) * 8;
    v8h lo = *(const v8h*)&Bt[n][bk];
    v8h hi = *(const v8h*)&Bt[n][bk + 16];
    return cat8(lo, hi);
}

__device__ inline float silu_fast(float g) {
    // g * rcp(1 + e^-g): v_exp_f32 + v_rcp_f32, no IEEE-division chain
    return g * __builtin_amdgcn_rcpf(1.0f + __expf(-g));
}

__device__ inline v4h cvt4(float4 f, float s) {
    v4h r = {(_Float16)(f.x * s), (_Float16)(f.y * s),
             (_Float16)(f.z * s), (_Float16)(f.w * s)};
    return r;
}

// ---------------------------------------------------------------------------
// Router: logits -> top-4 -> renormalized combine weights (dense [T, 16]).
// ---------------------------------------------------------------------------
__global__ __launch_bounds__(256)
void moe_router_kernel(const float* __restrict__ X,   // [T, D]
                       const float* __restrict__ Wr,  // [D, 16]
                       float* __restrict__ combine)   // [T, 16]
{
    __shared__ float wrs[256][16];

    const int t = blockIdx.x * 256 + threadIdx.x;

    float logit[NEXP];
#pragma unroll
    for (int e = 0; e < NEXP; ++e) logit[e] = 0.0f;

    for (int d0 = 0; d0 < DDIM; d0 += 256) {
#pragma unroll
        for (int i = 0; i < 16; ++i) {
            int e = threadIdx.x * 16 + i;
            wrs[e >> 4][e & 15] = Wr[(size_t)(d0 + (e >> 4)) * NEXP + (e & 15)];
        }
        __syncthreads();
        for (int d = 0; d < 256; ++d) {
            float xv = X[(size_t)t * DDIM + d0 + d];
#pragma unroll
            for (int e = 0; e < NEXP; ++e) logit[e] += xv * wrs[d][e];
        }
        __syncthreads();
    }

    float mx = logit[0];
#pragma unroll
    for (int e = 1; e < NEXP; ++e) mx = fmaxf(mx, logit[e]);

    unsigned mask = 0;
    int   idx[4];
    float wv[4];
    float csum = 0.0f;
#pragma unroll
    for (int k = 0; k < 4; ++k) {
        int best = 0; float bv = -3.4e38f;
#pragma unroll
        for (int e = 0; e < NEXP; ++e) {
            bool taken = (mask >> e) & 1u;
            if (!taken && logit[e] > bv) { bv = logit[e]; best = e; }
        }
        mask |= (1u << best);
        idx[k] = best;
        wv[k]  = __expf(logit[best] - mx);
        csum  += wv[k];
    }

    float cw[NEXP];
#pragma unroll
    for (int e = 0; e < NEXP; ++e) cw[e] = 0.0f;
#pragma unroll
    for (int k = 0; k < 4; ++k) cw[idx[k]] = wv[k] / csum;
#pragma unroll
    for (int e = 0; e < NEXP; ++e) combine[(size_t)t * NEXP + e] = cw[e];
}

// ---------------------------------------------------------------------------
// Fused gate/up GEMM + SiLU gating (pipelined, double-buffered):
//   H = silu(X @ (Wg*sg)) * (X @ (Wu*su)),  H stored as f16 [T, Fn]
// 256 threads = 8 waves in 4(M) x 2(N); block tile 128x64, K-step 32.
// Wave owns a 32x32 patch: 2 A frags + 4 B frags -> 8 WMMAs per K-step.
// ---------------------------------------------------------------------------
struct GUStage {
    float4 a[4];
    float4 g[2];
    float4 u[2];
};

__global__ __launch_bounds__(256)
void moe_gateup_kernel(const float* __restrict__ X,   // [T, Dk] f32
                       const float* __restrict__ Wg,  // [Dk, Fn] f32
                       const float* __restrict__ Wu,  // [Dk, Fn] f32
                       const float* sgp, const float* sup, int eIdx,
                       _Float16* __restrict__ H,      // [T, Fn]
                       int Dk, int Fn)
{
    __shared__ _Float16 As [2][TILE_M][TILE_K + KPAD];
    __shared__ _Float16 BgT[2][TILE_N][TILE_K + KPAD];
    __shared__ _Float16 BuT[2][TILE_N][TILE_K + KPAD];

    const int tid   = threadIdx.x;
    const int lane  = tid & 31;
    const int wave  = tid >> 5;
    const int wm    = wave >> 1;   // 0..3
    const int wn    = wave & 1;    // 0..1
    const int tBase = blockIdx.y * TILE_M;
    const int nBase = blockIdx.x * TILE_N;

    const float sg = sgp ? sgp[eIdx] : 1.0f;
    const float su = sup ? sup[eIdx] : 1.0f;

    // ---- register-staged tile loads (no LDS, no waits forced here) ----
    auto stage_load = [&](int k0, GUStage& st) {
#pragma unroll
        for (int i = 0; i < 4; ++i) {                 // A: 128x32 f32
            int idx = tid + i * 256;
            int r = idx >> 3, c4 = idx & 7;
            st.a[i] = *(const float4*)(X + (size_t)(tBase + r) * Dk + k0 + c4 * 4);
        }
#pragma unroll
        for (int i = 0; i < 2; ++i) {                 // B: 32x64 f32 x2
            int idx = tid + i * 256;
            int r = idx >> 4, c4 = idx & 15;
            size_t off = (size_t)(k0 + r) * Fn + nBase + c4 * 4;
            st.g[i] = *(const float4*)(Wg + off);
            st.u[i] = *(const float4*)(Wu + off);
        }
    };
    // ---- convert + LDS store (waits for the loads land here) ----
    auto stage_store = [&](const GUStage& st, int buf) {
#pragma unroll
        for (int i = 0; i < 4; ++i) {
            int idx = tid + i * 256;
            int r = idx >> 3, c4 = idx & 7;
            *(v4h*)&As[buf][r][c4 * 4] = cvt4(st.a[i], 1.0f);
        }
#pragma unroll
        for (int i = 0; i < 2; ++i) {
            int idx = tid + i * 256;
            int r = idx >> 4, c4 = idx & 15;
            v4h hg = cvt4(st.g[i], sg);
            v4h hu = cvt4(st.u[i], su);
            BgT[buf][c4 * 4 + 0][r] = hg[0];
            BgT[buf][c4 * 4 + 1][r] = hg[1];
            BgT[buf][c4 * 4 + 2][r] = hg[2];
            BgT[buf][c4 * 4 + 3][r] = hg[3];
            BuT[buf][c4 * 4 + 0][r] = hu[0];
            BuT[buf][c4 * 4 + 1][r] = hu[1];
            BuT[buf][c4 * 4 + 2][r] = hu[2];
            BuT[buf][c4 * 4 + 3][r] = hu[3];
        }
    };

    const v8f zero = {0.f, 0.f, 0.f, 0.f, 0.f, 0.f, 0.f, 0.f};
    v8f accG[2][2], accU[2][2];
#pragma unroll
    for (int mi = 0; mi < 2; ++mi)
#pragma unroll
        for (int ni = 0; ni < 2; ++ni) { accG[mi][ni] = zero; accU[mi][ni] = zero; }

    GUStage st;
    stage_load(0, st);
    stage_store(st, 0);
    __syncthreads();

    const int nsteps = Dk / TILE_K;
    for (int s = 0; s < nsteps; ++s) {
        const int  cur      = s & 1;
        const bool havenext = (s + 1 < nsteps);

        if (havenext) stage_load((s + 1) * TILE_K, st);   // loads in flight

        const int mr = lane & 15;
        v16h a[2], bg[2], bu[2];
#pragma unroll
        for (int mi = 0; mi < 2; ++mi)
            a[mi] = load_frag_a(As[cur], wm * 32 + mi * 16 + mr, lane);
#pragma unroll
        for (int ni = 0; ni < 2; ++ni) {
            bg[ni] = load_frag_bt(BgT[cur], wn * 32 + ni * 16 + mr, lane);
            bu[ni] = load_frag_bt(BuT[cur], wn * 32 + ni * 16 + mr, lane);
        }
#pragma unroll
        for (int mi = 0; mi < 2; ++mi)
#pragma unroll
            for (int ni = 0; ni < 2; ++ni) {
                accG[mi][ni] = __builtin_amdgcn_wmma_f32_16x16x32_f16(
                    false, a[mi], false, bg[ni], (short)0, accG[mi][ni], false, false);
                accU[mi][ni] = __builtin_amdgcn_wmma_f32_16x16x32_f16(
                    false, a[mi], false, bu[ni], (short)0, accU[mi][ni], false, false);
            }

        if (havenext) stage_store(st, cur ^ 1);           // waits land here
        __syncthreads();
    }

    // h = silu(gate) * up, written as f16
    const int row0 = tBase + wm * 32 + (lane >> 4) * 8;
    const int col0 = nBase + wn * 32 + (lane & 15);
#pragma unroll
    for (int mi = 0; mi < 2; ++mi)
#pragma unroll
        for (int ni = 0; ni < 2; ++ni)
#pragma unroll
            for (int v = 0; v < 8; ++v) {
                float g = accG[mi][ni][v];
                float u = accU[mi][ni][v];
                float h = silu_fast(g) * u;
                H[(size_t)(row0 + mi * 16 + v) * Fn + (col0 + ni * 16)] = (_Float16)h;
            }
}

// ---------------------------------------------------------------------------
// Down GEMM with per-row combine weight (pipelined, double-buffered):
//   Out[t, :] += alpha[t] * (H[t, :] @ (Wd*sd))
// alpha[t] = routedScale * combine[t, e]  (1.0 when combine == nullptr).
// Experts are launched sequentially -> deterministic accumulation.
// ---------------------------------------------------------------------------
struct DnStage {
    v4h    a[4];
    float4 w[2];
};

__global__ __launch_bounds__(256)
void moe_down_kernel(const _Float16* __restrict__ H, // [T, Fk] f16
                     const float* __restrict__ Wd,   // [Fk, Dn] f32
                     const float* sdp, int eIdx,
                     const float* __restrict__ combine, // [T, 16] or nullptr
                     float routedScale,
                     float* __restrict__ Out,        // [T, Dn]
                     int Fk, int Dn)
{
    __shared__ _Float16 As [2][TILE_M][TILE_K + KPAD];
    __shared__ _Float16 BdT[2][TILE_N][TILE_K + KPAD];
    __shared__ float alphaS[TILE_M];

    const int tid   = threadIdx.x;
    const int lane  = tid & 31;
    const int wave  = tid >> 5;
    const int wm    = wave >> 1;
    const int wn    = wave & 1;
    const int tBase = blockIdx.y * TILE_M;
    const int nBase = blockIdx.x * TILE_N;

    const float sd = sdp ? sdp[eIdx] : 1.0f;

    if (tid < TILE_M) {
        alphaS[tid] = combine
            ? routedScale * combine[(size_t)(tBase + tid) * NEXP + eIdx]
            : 1.0f;
    }

    auto stage_load = [&](int k0, DnStage& st) {
#pragma unroll
        for (int i = 0; i < 4; ++i) {                 // A from f16 H
            int idx = tid + i * 256;
            int r = idx >> 3, c4 = idx & 7;
            st.a[i] = *(const v4h*)(H + (size_t)(tBase + r) * Fk + k0 + c4 * 4);
        }
#pragma unroll
        for (int i = 0; i < 2; ++i) {                 // B: 32x64 f32
            int idx = tid + i * 256;
            int r = idx >> 4, c4 = idx & 15;
            st.w[i] = *(const float4*)(Wd + (size_t)(k0 + r) * Dn + nBase + c4 * 4);
        }
    };
    auto stage_store = [&](const DnStage& st, int buf) {
#pragma unroll
        for (int i = 0; i < 4; ++i) {
            int idx = tid + i * 256;
            int r = idx >> 3, c4 = idx & 7;
            *(v4h*)&As[buf][r][c4 * 4] = st.a[i];
        }
#pragma unroll
        for (int i = 0; i < 2; ++i) {
            int idx = tid + i * 256;
            int r = idx >> 4, c4 = idx & 15;
            v4h hw = cvt4(st.w[i], sd);
            BdT[buf][c4 * 4 + 0][r] = hw[0];
            BdT[buf][c4 * 4 + 1][r] = hw[1];
            BdT[buf][c4 * 4 + 2][r] = hw[2];
            BdT[buf][c4 * 4 + 3][r] = hw[3];
        }
    };

    const v8f zero = {0.f, 0.f, 0.f, 0.f, 0.f, 0.f, 0.f, 0.f};
    v8f acc[2][2];
#pragma unroll
    for (int mi = 0; mi < 2; ++mi)
#pragma unroll
        for (int ni = 0; ni < 2; ++ni) acc[mi][ni] = zero;

    DnStage st;
    stage_load(0, st);
    stage_store(st, 0);
    __syncthreads();

    const int nsteps = Fk / TILE_K;
    for (int s = 0; s < nsteps; ++s) {
        const int  cur      = s & 1;
        const bool havenext = (s + 1 < nsteps);

        if (havenext) stage_load((s + 1) * TILE_K, st);

        const int mr = lane & 15;
        v16h a[2], b[2];
#pragma unroll
        for (int mi = 0; mi < 2; ++mi)
            a[mi] = load_frag_a(As[cur], wm * 32 + mi * 16 + mr, lane);
#pragma unroll
        for (int ni = 0; ni < 2; ++ni)
            b[ni] = load_frag_bt(BdT[cur], wn * 32 + ni * 16 + mr, lane);
#pragma unroll
        for (int mi = 0; mi < 2; ++mi)
#pragma unroll
            for (int ni = 0; ni < 2; ++ni)
                acc[mi][ni] = __builtin_amdgcn_wmma_f32_16x16x32_f16(
                    false, a[mi], false, b[ni], (short)0, acc[mi][ni], false, false);

        if (havenext) stage_store(st, cur ^ 1);
        __syncthreads();
    }

    const int mrow = wm * 32 + (lane >> 4) * 8;
    const int col0 = nBase + wn * 32 + (lane & 15);
#pragma unroll
    for (int mi = 0; mi < 2; ++mi)
#pragma unroll
        for (int ni = 0; ni < 2; ++ni)
#pragma unroll
            for (int v = 0; v < 8; ++v) {
                int mloc = mrow + mi * 16 + v;
                size_t o = (size_t)(tBase + mloc) * Dn + (col0 + ni * 16);
                Out[o] += alphaS[mloc] * acc[mi][ni][v];
            }
}

// ---------------------------------------------------------------------------
// Host-side launch.
// ---------------------------------------------------------------------------
extern "C" void kernel_launch(void* const* d_in, const int* in_sizes, int n_in,
                              void* d_out, int out_size, void* d_ws, size_t ws_size,
                              hipStream_t stream) {
    const float* X   = (const float*)d_in[0];   // [T, D]
    const float* Wr  = (const float*)d_in[1];   // [D, 16]
    const float* Wg  = (const float*)d_in[2];   // [E, D, F]
    const float* Wu  = (const float*)d_in[3];   // [E, D, F]
    const float* Wd  = (const float*)d_in[4];   // [E, F, D]
    const float* Sg  = (const float*)d_in[5];   // [E]
    const float* Su  = (const float*)d_in[6];   // [E]
    const float* Sd  = (const float*)d_in[7];   // [E]
    const float* SWg = (const float*)d_in[8];   // [D, FS]
    const float* SWu = (const float*)d_in[9];   // [D, FS]
    const float* SWd = (const float*)d_in[10];  // [FS, D]
    float* Out = (float*)d_out;

    // workspace layout: combine [T,16] f32 | h buffer [T, max(F,FS)] f16
    float*    combine = (float*)d_ws;
    _Float16* hbuf    = (_Float16*)((char*)d_ws + (size_t)T_TOK * NEXP * sizeof(float));

    hipMemsetAsync(d_out, 0, (size_t)T_TOK * DDIM * sizeof(float), stream);

    moe_router_kernel<<<T_TOK / 256, 256, 0, stream>>>(X, Wr, combine);

    // shared expert (all tokens, weight 1.0)
    {
        dim3 gu(FSDIM / TILE_N, T_TOK / TILE_M);
        moe_gateup_kernel<<<gu, 256, 0, stream>>>(
            X, SWg, SWu, nullptr, nullptr, 0, hbuf, DDIM, FSDIM);
        dim3 dn(DDIM / TILE_N, T_TOK / TILE_M);
        moe_down_kernel<<<dn, 256, 0, stream>>>(
            hbuf, SWd, nullptr, 0, nullptr, 1.0f, Out, FSDIM, DDIM);
    }

    // routed experts, sequential over e (matches reference scan; deterministic)
    for (int e = 0; e < NEXP; ++e) {
        dim3 gu(FDIM / TILE_N, T_TOK / TILE_M);
        moe_gateup_kernel<<<gu, 256, 0, stream>>>(
            X, Wg + (size_t)e * DDIM * FDIM, Wu + (size_t)e * DDIM * FDIM,
            Sg, Su, e, hbuf, DDIM, FDIM);
        dim3 dn(DDIM / TILE_N, T_TOK / TILE_M);
        moe_down_kernel<<<dn, 256, 0, stream>>>(
            hbuf, Wd + (size_t)e * FDIM * DDIM, Sd, e,
            combine, 2.5f, Out, FDIM, DDIM);
    }
}
